// LocalAttention_20650202759981
// MI455X (gfx1250) — matrix operand order repaired and checked by reading, not compiled, not run on hardware
//
#include <hip/hip_runtime.h>

// ---------------------------------------------------------------------------
// Local window attention for MI455X (gfx1250): all matmuls via
// v_wmma_f32_16x16x32_bf16 (wave32 WMMA), bf16 intermediates, f32 accumulate.
// GEMM: 32x64 tile per wave (8 WMMA per K-step, B-frags reused x2),
// global_prefetch of the next K slice.
// ---------------------------------------------------------------------------

typedef __attribute__((ext_vector_type(16))) __bf16 v16bf;
typedef __attribute__((ext_vector_type(8)))  __bf16 bf16x8;
typedef __attribute__((ext_vector_type(8)))  float  v8f;

constexpr int D_MODEL   = 512;
constexpr int NUM_HEADS = 8;
constexpr int DH        = 64;          // head dim
constexpr int HALF_WIN  = 32;          // window = 65
constexpr int SEQ       = 4096;
constexpr int BATCH     = 2;
constexpr int LDQKV     = 3 * D_MODEL; // 1536

static __device__ __forceinline__ v8f wmma_bf16(v16bf a, v16bf b, v8f c) {
  return __builtin_amdgcn_wmma_f32_16x16x32_bf16(false, a, false, b,
                                                 (short)0, c, false, false);
}

// A-fragment (M=16, K=32, bf16) in this code's k order (not the ISA table order):
// lanes 0-15 hold row M=lane, K 0..15; lanes 16-31 K 16..31 (contiguous).
static __device__ __forceinline__ v16bf load_afrag(const __bf16* __restrict__ base,
                                                   int ld, int row0, int k0, int lane) {
  const int sub = lane & 15;
  const int hi8 = (lane >> 4) << 4;          // 0 or 16
  const __bf16* p = base + (size_t)(row0 + sub) * ld + k0 + hi8;
  bf16x8 lo = *(const bf16x8*)p;
  bf16x8 hi = *(const bf16x8*)(p + 8);
  v16bf f;
#pragma unroll
  for (int i = 0; i < 8; ++i) { f[i] = lo[i]; f[i + 8] = hi[i]; }
  return f;
}

// B-fragment (K=32, N=16, bf16): column n comes from source row (n0+n),
// lanes 0-15 hold K=0..15, lanes 16-31 hold K=16..31 (contiguous in memory).
static __device__ __forceinline__ v16bf load_bfrag(const __bf16* __restrict__ base,
                                                   int ld, int n0, int k0, int lane) {
  const int n    = lane & 15;
  const int koff = k0 + ((lane >> 4) << 4);  // +0 or +16
  const __bf16* p = base + (size_t)(n0 + n) * ld + koff;
  bf16x8 lo = *(const bf16x8*)p;
  bf16x8 hi = *(const bf16x8*)(p + 8);
  v16bf f;
#pragma unroll
  for (int i = 0; i < 8; ++i) { f[i] = lo[i]; f[i + 8] = hi[i]; }
  return f;
}

// ---------------------------------------------------------------------------
// f32 -> bf16 conversion
// ---------------------------------------------------------------------------
__global__ void f32_to_bf16_kernel(const float* __restrict__ in,
                                   __bf16* __restrict__ out, int n) {
  int i = blockIdx.x * blockDim.x + threadIdx.x;
  if (i < n) out[i] = (__bf16)in[i];
}

// ---------------------------------------------------------------------------
// C[M,N] = A[M,K] (bf16) @ W[N,K]^T (bf16) + bias[N]  ; C bf16 or f32.
// Block: 256 threads = 8 waves. Wave w -> rows m0 = by*256 + 32w (32 rows),
// cols 64*bx. Per K-step: 2 A-frags + 4 B-frags -> 8 WMMAs.
// ---------------------------------------------------------------------------
template <bool OUT_BF16>
__global__ __launch_bounds__(256)
void gemm_bf16_wt(const __bf16* __restrict__ A, const __bf16* __restrict__ W,
                  const float* __restrict__ bias, void* __restrict__ Cout,
                  int M, int N, int K) {
  const int lane = threadIdx.x & 31;
  const int wave = threadIdx.x >> 5;
  const int m0 = blockIdx.y * 256 + wave * 32;
  const int n0 = blockIdx.x * 64;

  v8f acc[2][4];
#pragma unroll
  for (int g = 0; g < 2; ++g)
#pragma unroll
    for (int t = 0; t < 4; ++t) acc[g][t] = {};

  for (int kk = 0; kk < K; kk += 32) {
    if (kk + 32 < K) {
      // hide L2->L0 latency for the streaming operands (global_prefetch_b8)
      __builtin_prefetch(A + (size_t)(m0 + (lane & 15)) * K + kk + 32, 0, 1);
      __builtin_prefetch(W + (size_t)(n0 + (lane & 15)) * K + kk + 32, 0, 1);
    }
    v16bf a0 = load_afrag(A, K, m0,      kk, lane);
    v16bf a1 = load_afrag(A, K, m0 + 16, kk, lane);
#pragma unroll
    for (int t = 0; t < 4; ++t) {
      v16bf b = load_bfrag(W, K, n0 + 16 * t, kk, lane);
      acc[0][t] = wmma_bf16(a0, b, acc[0][t]);
      acc[1][t] = wmma_bf16(a1, b, acc[1][t]);
    }
  }

  const int hi  = lane >> 4;
  const int col = lane & 15;
#pragma unroll
  for (int g = 0; g < 2; ++g) {
#pragma unroll
    for (int t = 0; t < 4; ++t) {
      const int c = n0 + 16 * t + col;
      const float bv = bias[c];
#pragma unroll
      for (int r = 0; r < 8; ++r) {
        const int row = m0 + 16 * g + r + hi * 8;
        const float v = acc[g][t][r] + bv;
        if constexpr (OUT_BF16)
          ((__bf16*)Cout)[(size_t)row * N + c] = (__bf16)v;
        else
          ((float*)Cout)[(size_t)row * N + c] = v;
      }
    }
  }
}

// ---------------------------------------------------------------------------
// Local attention: one wave per (b, h, 16-query tile).
// qkv: [B*S, 1536] bf16 (q|k|v).  ctx: [B*S, 512] bf16.
// ---------------------------------------------------------------------------
__global__ __launch_bounds__(32)
void local_attn_kernel(const __bf16* __restrict__ qkv, __bf16* __restrict__ ctx) {
  __shared__ alignas(16) float  sc[16][80];     // masked scores (f32)
  __shared__ alignas(16) __bf16 attnb[16][96];  // softmax probs, K-padded to 96
  __shared__ alignas(16) __bf16 vt[64][96];     // V band, transposed dh-major

  const int lane = threadIdx.x;
  const int sub  = lane & 15;
  const int hi   = lane >> 4;

  const int s0 = blockIdx.x * 16;
  const int b  = blockIdx.y / NUM_HEADS;
  const int h  = blockIdx.y % NUM_HEADS;
  const __bf16* rowbase = qkv + (size_t)b * SEQ * LDQKV;
  const int kbase = s0 - HALF_WIN;              // first key of the 80-key band

  // ---- Q fragments (two dh chunks of 32) -------------------------------
  v16bf qa[2];
#pragma unroll
  for (int c = 0; c < 2; ++c)
    qa[c] = load_afrag(rowbase, LDQKV, s0, h * DH + 32 * c, lane);

  // ---- scores: 5 key tiles of 16, 2 WMMAs each -------------------------
#pragma unroll
  for (int j = 0; j < 5; ++j) {
    const int p = kbase + 16 * j + sub;         // this lane's key row
    v16bf kb[2];
#pragma unroll
    for (int c = 0; c < 2; ++c) {
      v16bf f = {};
      if (p >= 0 && p < SEQ) {
        const __bf16* kp = rowbase + (size_t)p * LDQKV + D_MODEL + h * DH
                         + 32 * c + hi * 16;
        bf16x8 lo = *(const bf16x8*)kp;
        bf16x8 hh = *(const bf16x8*)(kp + 8);
#pragma unroll
        for (int i = 0; i < 8; ++i) { f[i] = lo[i]; f[i + 8] = hh[i]; }
      }
      kb[c] = f;
    }
    v8f acc = {};
    acc = wmma_bf16(qa[0], kb[0], acc);
    acc = wmma_bf16(qa[1], kb[1], acc);

    const int jj = 16 * j + sub;
#pragma unroll
    for (int r = 0; r < 8; ++r) {
      const int m   = r + hi * 8;
      const int rel = jj - m;                   // in-window iff 0..64
      const int pk  = kbase + jj;
      const bool ok = (rel >= 0) && (rel <= 2 * HALF_WIN) && (pk >= 0) && (pk < SEQ);
      sc[m][jj] = ok ? acc[r] * 0.125f : -3.0e38f;   // 0.125 = 1/sqrt(64)
    }
  }
  __syncthreads();

  // ---- stage V band transposed: vt[dh][key] ----------------------------
  for (int t = lane; t < 96 * 8; t += 32) {
    const int jj = t >> 3;                      // key column 0..95
    const int c8 = t & 7;                       // dh octet
    const int p  = kbase + jj;
    bf16x8 vals;
    if (jj < 80 && p >= 0 && p < SEQ) {
      vals = *(const bf16x8*)(rowbase + (size_t)p * LDQKV + 2 * D_MODEL
                              + h * DH + c8 * 8);
    } else {
#pragma unroll
      for (int u = 0; u < 8; ++u) vals[u] = (__bf16)0.0f;
    }
#pragma unroll
    for (int u = 0; u < 8; ++u) vt[c8 * 8 + u][jj] = vals[u];
  }

  // ---- softmax, one row per lane (lanes 0..15) -------------------------
  if (lane < 16) {
    float mx = -3.0e38f;
    for (int jj = 0; jj < 80; ++jj) mx = fmaxf(mx, sc[lane][jj]);
    float sum = 0.0f;
    for (int jj = 0; jj < 80; ++jj) {
      float e = __expf(sc[lane][jj] - mx);
      sc[lane][jj] = e;
      sum += e;
    }
    const float inv = 1.0f / sum;
    for (int jj = 0; jj < 96; ++jj)
      attnb[lane][jj] = (__bf16)((jj < 80) ? sc[lane][jj] * inv : 0.0f);
  }
  __syncthreads();

  // ---- ctx = attn(16x96) @ V(96x64): 3 K-chunks x 4 dh tiles -----------
  v16bf af[3];
#pragma unroll
  for (int c = 0; c < 3; ++c) {
    const __bf16* p = &attnb[sub][32 * c + hi * 16];
    bf16x8 lo = *(const bf16x8*)p;
    bf16x8 hh = *(const bf16x8*)(p + 8);
#pragma unroll
    for (int i = 0; i < 8; ++i) { af[c][i] = lo[i]; af[c][i + 8] = hh[i]; }
  }

#pragma unroll
  for (int t = 0; t < 4; ++t) {
    v8f acc = {};
#pragma unroll
    for (int c = 0; c < 3; ++c) {
      const __bf16* p = &vt[16 * t + sub][32 * c + hi * 16];
      bf16x8 lo = *(const bf16x8*)p;
      bf16x8 hh = *(const bf16x8*)(p + 8);
      v16bf bfr;
#pragma unroll
      for (int i = 0; i < 8; ++i) { bfr[i] = lo[i]; bfr[i + 8] = hh[i]; }
      acc = wmma_bf16(af[c], bfr, acc);
    }
#pragma unroll
    for (int r = 0; r < 8; ++r) {
      const int row = s0 + r + hi * 8;
      const int col = h * DH + 16 * t + sub;
      ctx[((size_t)b * SEQ + row) * D_MODEL + col] = (__bf16)acc[r];
    }
  }
}

// ---------------------------------------------------------------------------
// Host launcher
// ---------------------------------------------------------------------------
extern "C" void kernel_launch(void* const* d_in, const int* in_sizes, int n_in,
                              void* d_out, int out_size, void* d_ws, size_t ws_size,
                              hipStream_t stream) {
  (void)in_sizes; (void)n_in; (void)out_size; (void)ws_size;

  const float* x     = (const float*)d_in[0];
  const float* w_in  = (const float*)d_in[1];
  const float* b_in  = (const float*)d_in[2];
  const float* w_out = (const float*)d_in[3];
  const float* b_out = (const float*)d_in[4];

  char* ws = (char*)d_ws;
  size_t off = 0;
  auto carve = [&](size_t bytes) {
    char* p = ws + off;
    off += (bytes + 255) & ~(size_t)255;
    return p;
  };

  const int M = BATCH * SEQ;                       // 8192
  __bf16* x_bf    = (__bf16*)carve((size_t)M * D_MODEL * 2);
  __bf16* win_bf  = (__bf16*)carve((size_t)3 * D_MODEL * D_MODEL * 2);
  __bf16* wout_bf = (__bf16*)carve((size_t)D_MODEL * D_MODEL * 2);
  __bf16* qkv_bf  = (__bf16*)carve((size_t)M * LDQKV * 2);
  __bf16* ctx_bf  = (__bf16*)carve((size_t)M * D_MODEL * 2);

  const int nx  = M * D_MODEL;
  const int nwi = 3 * D_MODEL * D_MODEL;
  const int nwo = D_MODEL * D_MODEL;
  f32_to_bf16_kernel<<<(nx  + 255) / 256, 256, 0, stream>>>(x,     x_bf,    nx);
  f32_to_bf16_kernel<<<(nwi + 255) / 256, 256, 0, stream>>>(w_in,  win_bf,  nwi);
  f32_to_bf16_kernel<<<(nwo + 255) / 256, 256, 0, stream>>>(w_out, wout_bf, nwo);

  // qkv = x @ in_proj_w^T + in_proj_b   (bf16 out)
  dim3 g1(3 * D_MODEL / 64, M / 256);
  gemm_bf16_wt<true><<<g1, 256, 0, stream>>>(x_bf, win_bf, b_in, qkv_bf,
                                             M, 3 * D_MODEL, D_MODEL);

  // sliding-window attention -> ctx (bf16)
  dim3 g2(SEQ / 16, BATCH * NUM_HEADS);
  local_attn_kernel<<<g2, 32, 0, stream>>>(qkv_bf, ctx_bf);

  // out = ctx @ out_proj_w^T + out_proj_b  (f32 out)
  dim3 g3(D_MODEL / 64, M / 256);
  gemm_bf16_wt<false><<<g3, 256, 0, stream>>>(ctx_bf, wout_bf, b_out,
                                              (float*)d_out, M, D_MODEL, D_MODEL);
}
